// TNNColumnLayer_67216238182820
// MI455X (gfx1250) — matrix operand order repaired and checked by reading, not compiled, not run on hardware
//
#include <hip/hip_runtime.h>

typedef __attribute__((ext_vector_type(16))) _Float16 v16h;
typedef __attribute__((ext_vector_type(8)))  float    v8f;

#define HH      64
#define WW      64
#define NPREV   8
#define RF      4
#define STRIDEC 2
#define QN      16
#define ROWSN   31
#define COLSN   31
#define PN      128
#define NUMN    (ROWSN*COLSN*QN)   /* 15376 */
#define THETA   80.0f
#define MAXT    72                 /* TMAX + WMAX + 1 */
#define KPAD    96                 /* 3 k-steps of 32 */
#define NPAD    80                 /* 5 n-tiles of 16 */

// One workgroup (one wave32) per spatial column.
// Phase 1: gather the 128 patch spike-times into LDS.
// Phase 2: lanes 0..15 build per-neuron signed event histograms A[16][96] (i16, LDS).
// Phase 3: single wave computes tpot[16][80] = A(16x96,f16) @ Ramp(96x80,f16) with
//          v_wmma_f32_16x16x32_f16; the ramp operand max(t+1-k,0) is synthesized per lane.
// Phase 4: threshold scan, per-column winner-take-all, stream outputs (float4).
__global__ __launch_bounds__(32)
void tnn_column_wmma_kernel(const float* __restrict__ data,      // (64,64,8)
                            const float* __restrict__ weights,   // (NUM, P)
                            float* __restrict__ out_next,        // NUM
                            float* __restrict__ out_inp,         // NUM*P
                            float* __restrict__ out_stdp)        // NUM*P
{
    __shared__ __align__(16) float tvals[PN];   // 16B-aligned: read back as float4 (b128)
    __shared__ short Ahist[QN][KPAD];
    __shared__ float tpot[QN][NPAD];
    __shared__ float ecs[QN];
    __shared__ float pots[QN];
    __shared__ int   winner_s;
    __shared__ float mins_s;

    const int col  = blockIdx.x;          // 0..960
    const int rr   = col / COLSN;
    const int cc   = col % COLSN;
    const int lane = threadIdx.x;         // 0..31
    const int m    = lane & 15;
    const int hi   = lane >> 4;
    const float INFV = __builtin_inff();

    // ---- zero histogram (incl. K padding 72..95) ----
    for (int i = lane; i < QN * KPAD; i += 32) (&Ahist[0][0])[i] = 0;

    // ---- phase 1: patch gather.  p = np*16 + i*4 + j  ->  data[r*2+i][c*2+j][np]
    for (int p = lane; p < PN; p += 32) {
        const int np = p >> 4;
        const int ij = p & 15;
        const int pi = ij >> 2;
        const int pj = ij & 3;
        tvals[p] = data[((rr * STRIDEC + pi) * WW + (cc * STRIDEC + pj)) * NPREV + np];
    }
    __syncthreads();

    // ---- phase 2: signed event histogram, one lane per neuron row ----
    if (lane < QN) {
        const float* wrow = weights + (size_t)(col * QN + lane) * PN;
        for (int p = 0; p < PN; ++p) {
            const float t = tvals[p];
            if (!isinf(t)) {                       // inf spikes: eff_w = 0, never active
                int r = (int)rintf(wrow[p]);       // round-half-even, matches jnp.round
                if (r < 0) r = 0;
                const int s0 = (int)t;             // integer times in [0, 64)
                Ahist[lane][s0] += 1;              // +relu ramp at arrival
                const int s1 = s0 + r;
                if (s1 < MAXT) Ahist[lane][s1] -= 1;   // -relu ramp at end (>=72 never matters)
            }
        }
    }
    __syncthreads();

    // ---- phase 3: tpot = A @ Ramp via WMMA (whole wave, EXEC all ones) ----
    v8f acc[5];
    #pragma unroll
    for (int nt = 0; nt < 5; ++nt) {
        v8f z = {0.f, 0.f, 0.f, 0.f, 0.f, 0.f, 0.f, 0.f};
        acc[nt] = z;
    }
    #pragma unroll
    for (int ks = 0; ks < 3; ++ks) {
        // A fragment: lane holds row m; half v <-> k per ISA 16-bit 16x32 layout
        v16h a;
        #pragma unroll
        for (int v = 0; v < 16; ++v) {
            const int k = ks * 32 + ((v < 8) ? (hi * 8 + v) : (16 + hi * 8 + (v - 8)));
            a[v] = (_Float16)(float)Ahist[m][k];
        }
        #pragma unroll
        for (int nt = 0; nt < 5; ++nt) {
            // B fragment: Ramp[k][n] = max(n+1-k, 0); lane holds column n = m
            v16h b;
            #pragma unroll
            for (int v = 0; v < 16; ++v) {
                const int k = ks * 32 + ((v < 8) ? (hi * 8 + v) : (16 + hi * 8 + (v - 8)));
                const int n = nt * 16 + m;
                const int x = n + 1 - k;
                b[v] = (_Float16)(float)(x > 0 ? x : 0);
            }
            acc[nt] = __builtin_amdgcn_wmma_f32_16x16x32_f16(
                false, a, false, b, (short)0, acc[nt], false, false);
        }
    }
    // C/D layout: lane -> column n = m, VGPR v -> row v + 8*hi
    #pragma unroll
    for (int nt = 0; nt < 5; ++nt)
        #pragma unroll
        for (int v = 0; v < 8; ++v)
            tpot[v + 8 * hi][nt * 16 + m] = acc[nt][v];
    __syncthreads();

    // ---- phase 4a: threshold scan per neuron ----
    if (lane < QN) {
        const float* row = tpot[lane];
        int tidx = -1;
        for (int t = 0; t < MAXT; ++t) {
            if (row[t] >= THETA) { tidx = t; break; }
        }
        if (tidx < 0) { ecs[lane] = INFV;        pots[lane] = row[0]; }
        else          { ecs[lane] = (float)tidx; pots[lane] = row[tidx]; }
    }
    __syncthreads();

    // ---- phase 4b: winner-take-all within the column ----
    if (lane == 0) {
        float mn = INFV;
        for (int q = 0; q < QN; ++q) mn = fminf(mn, ecs[q]);
        int   win  = 0;
        float best = -1.0f;
        for (int q = 0; q < QN; ++q) {
            const float score = (ecs[q] == mn) ? pots[q] : 0.0f;  // inf==inf -> all-null case ok
            if (score > best) { best = score; win = q; }          // first max, like jnp.argmax
        }
        winner_s = win;
        mins_s   = mn;
    }
    __syncthreads();

    // ---- phase 4c: outputs ----
    if (lane < QN)
        out_next[col * QN + lane] = (lane == winner_s) ? mins_s : INFV;

    float4* inp4  = (float4*)(out_inp  + (size_t)col * QN * PN);
    float4* stdp4 = (float4*)(out_stdp + (size_t)col * QN * PN);
    const int win = winner_s;
    const float mn = mins_s;
    for (int idx = lane; idx < QN * (PN / 4); idx += 32) {
        const int q = idx / (PN / 4);
        const int e = idx % (PN / 4);
        inp4[idx] = ((const float4*)tvals)[e];                 // same patch for all 16 rows
        const float lv = (q == win) ? mn : INFV;
        stdp4[idx] = make_float4(lv, lv, lv, lv);
    }
}

extern "C" void kernel_launch(void* const* d_in, const int* in_sizes, int n_in,
                              void* d_out, int out_size, void* d_ws, size_t ws_size,
                              hipStream_t stream) {
    (void)in_sizes; (void)n_in; (void)d_ws; (void)ws_size; (void)out_size;
    const float* data    = (const float*)d_in[0];   // (64,64,8) f32
    const float* weights = (const float*)d_in[1];   // (15376,128) f32
    float* out      = (float*)d_out;
    float* out_next = out;                                   // 15376
    float* out_inp  = out + NUMN;                            // 15376*128
    float* out_stdp = out + NUMN + (size_t)NUMN * PN;        // 15376*128
    tnn_column_wmma_kernel<<<ROWSN * COLSN, 32, 0, stream>>>(
        data, weights, out_next, out_inp, out_stdp);
}
